// Attention_13091060318555
// MI455X (gfx1250) — matrix operand order repaired and checked by reading, not compiled, not run on hardware
//
#include <hip/hip_runtime.h>

// ---------------------------------------------------------------------------
// MHA forward for MI455X (gfx1250), bf16 WMMA (16x16x32) with fp32 accum.
// B=4, S=2048, D=1024, H=16, hd=64, scale = 1/8.
// Round 2: async global->LDS tile staging (GLOBAL_LOAD_ASYNC_TO_LDS_B128 +
// s_wait_asynccnt) for the non-transposed tiles (GEMM A tile, attention K tile).
// ---------------------------------------------------------------------------

#define DEVI static __device__ __forceinline__

typedef __bf16 bf16x16 __attribute__((ext_vector_type(16)));
typedef float f32x8 __attribute__((ext_vector_type(8)));

union AFragU { bf16x16 v; unsigned int w[8]; };
union AccU { f32x8 v; float f[8]; };
union Vec16B { uint4 q; unsigned short h[8]; };

DEVI unsigned short f2bf(float x) {
  union { float f; unsigned int u; } c;
  c.f = x;
  unsigned int r = 0x7fffu + ((c.u >> 16) & 1u);  // round-to-nearest-even
  return (unsigned short)((c.u + r) >> 16);
}

// Async DMA one 16-byte chunk per lane: global -> LDS (no VGPR data path).
// lds_off: workgroup-relative LDS byte address (low 32 bits of generic ptr).
DEVI void async_copy_b128(const void* lds_ptr, const void* gptr) {
  unsigned lds_off = (unsigned)(uintptr_t)lds_ptr;
  unsigned long long ga = (unsigned long long)(uintptr_t)gptr;
  asm volatile("global_load_async_to_lds_b128 %0, %1, off"
               :
               : "v"(lds_off), "v"(ga)
               : "memory");
}

DEVI void async_wait0() {
  asm volatile("s_wait_asynccnt 0" ::: "memory");
}

// A fragment: logical 16x32 (MxK), source row-major (K contiguous), stride in elems.
// ISA layout: lanes 0-15 M=0..15 K={0..7,16..23}; lanes 16-31 K={8..15,24..31}.
DEVI bf16x16 load_a_frag(const unsigned short* base, int stride) {
  int lane = threadIdx.x & 31;
  int row = lane & 15;
  int koff = (lane >> 4) << 3;  // 0 or 8
  AFragU f;
#pragma unroll
  for (int v = 0; v < 8; ++v) {
    int k = ((v >> 2) << 4) + koff + ((v & 3) << 1);
    f.w[v] = *(const unsigned int*)(base + row * stride + k);
  }
  return f.v;
}

// B fragment: logical 32x16 (KxN), storage arr[n*stride + k] (K contiguous).
// ISA layout: lanes 0-15 N=0..15 K={0..7,16..23}; lanes 16-31 K={8..15,24..31}.
DEVI bf16x16 load_b_frag(const unsigned short* base, int stride) {
  int lane = threadIdx.x & 31;
  int n = lane & 15;
  int koff = (lane >> 4) << 3;  // 0 or 8
  AFragU f;
#pragma unroll
  for (int v = 0; v < 8; ++v) {
    f.w[v] = *(const unsigned int*)(base + n * stride + ((v >> 2) << 4) + koff + ((v & 3) << 1));
  }
  return f.v;
}

DEVI f32x8 wmma_bf16(bf16x16 a, bf16x16 b, f32x8 c) {
  return __builtin_amdgcn_wmma_f32_16x16x32_bf16(false, a, false, b, (short)0, c,
                                                 false, false);
}

// ---------------------------------------------------------------------------
// Kernel 1: fp32 -> bf16 convert (grid-stride)
// ---------------------------------------------------------------------------
__global__ void cvt_bf16_kernel(const float* __restrict__ in,
                                unsigned short* __restrict__ out, int n) {
  for (int i = blockIdx.x * blockDim.x + threadIdx.x; i < n;
       i += gridDim.x * blockDim.x)
    out[i] = f2bf(in[i]);
}

// ---------------------------------------------------------------------------
// Kernel 2/4: bf16 GEMM  C[M,N] = A[M,K] @ W[K,N] (+bias)
// Block: 256 thr (8 waves), tile 64(M) x 128(N), K-step 32.
// Wave grid 4x2; each wave computes 16x64 via 4 accumulators.
// mode 0: QKV epilogue (scatter bf16 q/k/v, scale q). mode 1: proj (fp32 out).
// ---------------------------------------------------------------------------
__global__ __launch_bounds__(256) void gemm_bf16_kernel(
    const unsigned short* __restrict__ A, const unsigned short* __restrict__ W,
    const float* __restrict__ bias, int K, int N, int mode,
    unsigned short* __restrict__ qb, unsigned short* __restrict__ kb,
    unsigned short* __restrict__ vb, float* __restrict__ out) {
  __shared__ unsigned short As[64 * 40];   // 64 rows x (32+8) pad
  __shared__ unsigned short Bs[128 * 40];  // 128 n-rows x (32+8), K contiguous

  const int t = threadIdx.x;
  const int lane = t & 31;
  const int wid = t >> 5;
  const int waveM = wid >> 1;  // 0..3
  const int waveN = wid & 1;   // 0..1
  const int bM = blockIdx.y * 64;
  const int bN = blockIdx.x * 128;
  const int hl = lane >> 4;
  const int ln = lane & 15;

  AccU acc[4];
#pragma unroll
  for (int nt = 0; nt < 4; ++nt)
#pragma unroll
    for (int r = 0; r < 8; ++r) acc[nt].f[r] = 0.0f;

  const int ar = t >> 2;             // 0..63 A tile row
  const int ac = (t & 3) << 3;       // 0,8,16,24

  for (int kt = 0; kt < K; kt += 32) {
    // ---- A tile: async DMA global -> LDS (16B per lane, no VGPR data) ----
    async_copy_b128(&As[ar * 40 + ac], &A[(size_t)(bM + ar) * K + kt + ac]);
    if (kt + 32 < K)  // gfx1250 global_prefetch_b8 for next K tile
      __builtin_prefetch(&A[(size_t)(bM + ar) * K + kt + 32 + ac], 0, 1);
    // ---- B tile, transposed into K-contiguous storage (2 uint4 per thread) ----
#pragma unroll
    for (int i = 0; i < 2; ++i) {
      int cid = t * 2 + i;            // 0..511
      int kr = cid >> 4;              // 0..31
      int nc = (cid & 15) << 3;       // 0..120
      Vec16B u;
      u.q = *(const uint4*)&W[(size_t)(kt + kr) * N + bN + nc];
#pragma unroll
      for (int j = 0; j < 8; ++j) Bs[(nc + j) * 40 + kr] = u.h[j];
    }
    async_wait0();
    __syncthreads();

    bf16x16 af = load_a_frag(&As[waveM * 16 * 40], 40);
#pragma unroll
    for (int nt = 0; nt < 4; ++nt) {
      bf16x16 bf = load_b_frag(&Bs[(waveN * 64 + nt * 16) * 40], 40);
      acc[nt].v = wmma_bf16(af, bf, acc[nt].v);
    }
    __syncthreads();
  }

  // ---- epilogue ----
#pragma unroll
  for (int nt = 0; nt < 4; ++nt) {
#pragma unroll
    for (int r = 0; r < 8; ++r) {
      int m_g = bM + waveM * 16 + r + 8 * hl;
      int n_g = bN + waveN * 64 + nt * 16 + ln;
      float val = acc[nt].f[r] + bias[n_g];
      if (mode == 0) {
        // scatter into q/k/v [B,H,S,hd] bf16; fold 1/sqrt(hd) into q
        int which = n_g >> 10;
        int rem = n_g & 1023;
        int h = rem >> 6, c = rem & 63;
        int bi = m_g >> 11, s = m_g & 2047;
        size_t di = ((size_t)(bi * 16 + h) * 2048 + s) * 64 + c;
        if (which == 0)
          qb[di] = f2bf(val * 0.125f);
        else if (which == 1)
          kb[di] = f2bf(val);
        else
          vb[di] = f2bf(val);
      } else {
        out[(size_t)m_g * 1024 + n_g] = val;
      }
    }
  }
}

// ---------------------------------------------------------------------------
// Kernel 3: flash attention over [B*H] x (S/128) query slabs.
// 8 waves/block, each wave owns 16 query rows; stream 64-key tiles.
// q/k/v: bf16 [B*H, S, 64]; out ab: bf16 [B*S, 1024] (d = h*64 + c).
// ---------------------------------------------------------------------------
__global__ __launch_bounds__(256) void attn_kernel(
    const unsigned short* __restrict__ qb, const unsigned short* __restrict__ kb,
    const unsigned short* __restrict__ vb, unsigned short* __restrict__ ab) {
  __shared__ unsigned short Ks[64 * 72];      // [key][hd]   (hd contiguous)
  __shared__ unsigned short Vt[64 * 72];      // [hd][key]   (key contiguous)
  __shared__ unsigned short Ps[8 * 16 * 72];  // per-wave P staging 16x64

  const int t = threadIdx.x;
  const int lane = t & 31;
  const int wid = t >> 5;
  const int hl = lane >> 4;
  const int ln = lane & 15;
  const int bh = blockIdx.x;          // 0..63
  const int qbase = blockIdx.y * 128; // 16 slabs

  // Q fragments straight from global (row-major, K contiguous)
  const unsigned short* qrow =
      qb + (size_t)(bh * 2048 + qbase + wid * 16) * 64;
  bf16x16 qf0 = load_a_frag(qrow, 64);
  bf16x16 qf1 = load_a_frag(qrow + 32, 64);

  AccU o[4];
  float mrow[8], lrow[8];
#pragma unroll
  for (int nt = 0; nt < 4; ++nt)
#pragma unroll
    for (int r = 0; r < 8; ++r) o[nt].f[r] = 0.0f;
#pragma unroll
  for (int r = 0; r < 8; ++r) {
    mrow[r] = -__builtin_inff();
    lrow[r] = 0.0f;
  }

  unsigned short* Pw = &Ps[wid * 16 * 72];

  for (int kt0 = 0; kt0 < 2048; kt0 += 64) {
    const unsigned short* Kg = kb + (size_t)(bh * 2048 + kt0) * 64;
    const unsigned short* Vg = vb + (size_t)(bh * 2048 + kt0) * 64;
    // ---- K tile: async DMA global -> LDS; V tile: VGPR transpose path ----
#pragma unroll
    for (int i = 0; i < 2; ++i) {
      int cid = t * 2 + i;            // 0..511
      int key = cid >> 3;             // 0..63
      int hdo = (cid & 7) << 3;       // 0..56
      async_copy_b128(&Ks[key * 72 + hdo], &Kg[key * 64 + hdo]);
      Vec16B u;
      u.q = *(const uint4*)&Vg[key * 64 + hdo];
#pragma unroll
      for (int j = 0; j < 8; ++j) Vt[(hdo + j) * 72 + key] = u.h[j];
    }
    async_wait0();
    __syncthreads();

    // ---- S = Q @ K^T  (scale pre-folded into Q) ----
    AccU s[4];
#pragma unroll
    for (int nt = 0; nt < 4; ++nt)
#pragma unroll
      for (int r = 0; r < 8; ++r) s[nt].f[r] = 0.0f;
#pragma unroll
    for (int nt = 0; nt < 4; ++nt) {
      bf16x16 b0 = load_b_frag(&Ks[nt * 16 * 72 + 0], 72);
      s[nt].v = wmma_bf16(qf0, b0, s[nt].v);
      bf16x16 b1 = load_b_frag(&Ks[nt * 16 * 72 + 32], 72);
      s[nt].v = wmma_bf16(qf1, b1, s[nt].v);
    }

    // ---- online softmax: rows m = r + 8*hl, N across lanes of half-wave ----
    float corr[8];
#pragma unroll
    for (int r = 0; r < 8; ++r) {
      float mx = s[0].f[r];
      mx = fmaxf(mx, s[1].f[r]);
      mx = fmaxf(mx, s[2].f[r]);
      mx = fmaxf(mx, s[3].f[r]);
#pragma unroll
      for (int d = 1; d < 16; d <<= 1) mx = fmaxf(mx, __shfl_xor(mx, d, 32));
      float mnew = fmaxf(mrow[r], mx);
      float c = __expf(mrow[r] - mnew);
      float sum = 0.0f;
#pragma unroll
      for (int nt = 0; nt < 4; ++nt) {
        float p = __expf(s[nt].f[r] - mnew);
        s[nt].f[r] = p;
        sum += p;
      }
#pragma unroll
      for (int d = 1; d < 16; d <<= 1) sum += __shfl_xor(sum, d, 32);
      lrow[r] = lrow[r] * c + sum;
      mrow[r] = mnew;
      corr[r] = c;
    }
#pragma unroll
    for (int nt = 0; nt < 4; ++nt)
#pragma unroll
      for (int r = 0; r < 8; ++r) o[nt].f[r] *= corr[r];

    // ---- stage P (D layout -> row-major) through wave-private LDS ----
#pragma unroll
    for (int nt = 0; nt < 4; ++nt)
#pragma unroll
      for (int r = 0; r < 8; ++r)
        Pw[(r + 8 * hl) * 72 + nt * 16 + ln] = f2bf(s[nt].f[r]);
    __syncthreads();

    // ---- O += P @ V ----
    bf16x16 pa0 = load_a_frag(Pw, 72);
    bf16x16 pa1 = load_a_frag(Pw + 32, 72);
#pragma unroll
    for (int nt = 0; nt < 4; ++nt) {
      bf16x16 bv0 = load_b_frag(&Vt[nt * 16 * 72 + 0], 72);
      o[nt].v = wmma_bf16(pa0, bv0, o[nt].v);
      bf16x16 bv1 = load_b_frag(&Vt[nt * 16 * 72 + 32], 72);
      o[nt].v = wmma_bf16(pa1, bv1, o[nt].v);
    }
    __syncthreads();
  }

  // ---- normalize + write bf16 attention output [B*S, 1024] ----
  const int b = bh >> 4, h = bh & 15;
#pragma unroll
  for (int r = 0; r < 8; ++r) {
    float inv = 1.0f / lrow[r];
    int m = r + 8 * hl;
    size_t row = (size_t)(b * 2048 + qbase + wid * 16 + m);
#pragma unroll
    for (int nt = 0; nt < 4; ++nt)
      ab[row * 1024 + h * 64 + nt * 16 + ln] = f2bf(o[nt].f[r] * inv);
  }
}

// ---------------------------------------------------------------------------
// Host-side launcher
// ---------------------------------------------------------------------------
extern "C" void kernel_launch(void* const* d_in, const int* in_sizes, int n_in,
                              void* d_out, int out_size, void* d_ws,
                              size_t ws_size, hipStream_t stream) {
  (void)in_sizes; (void)n_in; (void)out_size; (void)ws_size;
  const float* x = (const float*)d_in[0];       // [4,2048,1024]
  const float* w_qkv = (const float*)d_in[1];   // [1024,3072]
  const float* b_qkv = (const float*)d_in[2];   // [3072]
  const float* w_proj = (const float*)d_in[3];  // [1024,1024]
  const float* b_proj = (const float*)d_in[4];  // [1024]
  float* out = (float*)d_out;                   // [4,2048,1024]

  const int M = 8192, K = 1024;
  char* ws = (char*)d_ws;
  size_t off = 0;
  unsigned short* xb = (unsigned short*)(ws + off); off += (size_t)M * K * 2;
  unsigned short* wqkvb = (unsigned short*)(ws + off); off += (size_t)K * 3072 * 2;
  unsigned short* wprojb = (unsigned short*)(ws + off); off += (size_t)K * 1024 * 2;
  unsigned short* qb = (unsigned short*)(ws + off); off += (size_t)M * 1024 * 2;
  unsigned short* kb = (unsigned short*)(ws + off); off += (size_t)M * 1024 * 2;
  unsigned short* vb = (unsigned short*)(ws + off); off += (size_t)M * 1024 * 2;
  unsigned short* ab = (unsigned short*)(ws + off); off += (size_t)M * 1024 * 2;

  cvt_bf16_kernel<<<4096, 256, 0, stream>>>(x, xb, M * K);
  cvt_bf16_kernel<<<2048, 256, 0, stream>>>(w_qkv, wqkvb, K * 3072);
  cvt_bf16_kernel<<<1024, 256, 0, stream>>>(w_proj, wprojb, K * 1024);

  gemm_bf16_kernel<<<dim3(3072 / 128, M / 64), 256, 0, stream>>>(
      xb, wqkvb, b_qkv, K, 3072, /*mode=*/0, qb, kb, vb, nullptr);

  attn_kernel<<<dim3(64, 16), 256, 0, stream>>>(qb, kb, vb, ab);

  gemm_bf16_kernel<<<dim3(1024 / 128, M / 64), 256, 0, stream>>>(
      ab, wprojb, b_proj, K, 1024, /*mode=*/1, nullptr, nullptr, nullptr, out);
}
